// Actor_55113020342523
// MI455X (gfx1250) — matrix operand order repaired and checked by reading, not compiled, not run on hardware
//
#include <hip/hip_runtime.h>
#include <hip/hip_bf16.h>
#include <math.h>

// ---------------- problem constants ----------------
#define OBS 128
#define ACTD 8
#define TT  256
#define BB  1024
#define H1  512
#define H2  256
#define HL  128
#define TBN (TT*BB)            // 262144

#define LOG2PI 1.8378770664093453f

// output layout (floats): action | logprob | entropy | hT | cT
#define OUT_LP  ((size_t)TBN*ACTD)          // 2097152
#define OUT_ENT (OUT_LP + (size_t)TBN)      // 2359296
#define OUT_H   (OUT_ENT + (size_t)TBN)     // 2621440
#define OUT_C   (OUT_H + (size_t)BB*HL)     // 2752512

typedef __attribute__((ext_vector_type(16))) _Float16 v16h;
typedef __attribute__((ext_vector_type(8)))  _Float16 v8h;
typedef __attribute__((ext_vector_type(8)))  float    v8f;

// ---------------- WMMA helpers ----------------
__device__ __forceinline__ v8f wmma_f16(v16h a, v16h b, v8f c) {
  // D = A(16x32 f16) * B(32x16 f16) + C(16x16 f32)
  return __builtin_amdgcn_wmma_f32_16x16x32_f16(
      /*neg_a*/false, a, /*neg_b*/false, b,
      /*c_mod*/(short)0, c, /*reuse_a*/false, /*reuse_b*/false);
}

// A-matrix fragment (16x32, f16): lane L holds row M=L&15; half=L>>4.
// VGPR v holds K pair: k = (v<4 ? 2v : 8+2v) + 8*half  (per ISA 16-bit A layout)
template<bool F32SRC>
__device__ __forceinline__ v16h load_a_g(const void* A, int ld, int m0, int k0) {
  int lane = threadIdx.x & 31;
  int half = lane >> 4;
  int m = m0 + (lane & 15);
  union { v16h v; _Float16 h[16]; } u;
  if (F32SRC) {
    const float* row = (const float*)A + (size_t)m * ld + k0 + half * 8;
#pragma unroll
    for (int p = 0; p < 8; ++p) {
      int kk = (p < 4) ? 2 * p : 8 + 2 * p;
      float2 f = *(const float2*)(row + kk);
      u.h[2 * p]     = (_Float16)f.x;
      u.h[2 * p + 1] = (_Float16)f.y;
    }
  } else {
    const _Float16* row = (const _Float16*)A + (size_t)m * ld + k0 + half * 8;
#pragma unroll
    for (int p = 0; p < 8; ++p) {
      int kk = (p < 4) ? 2 * p : 8 + 2 * p;
      u.h[2 * p]     = row[kk];
      u.h[2 * p + 1] = row[kk + 1];
    }
  }
  return u.v;
}

// B-matrix fragment (32x16, f16) from row-major f16 weight W[N][K] (B = W^T):
// lane L: column n = L&15; elems 0..7 <-> k = k0 + 8*(L>>4) + e, elems 8..15 <-> k = k0 + 16 + 8*(L>>4) + (e-8)
__device__ __forceinline__ v16h load_b_gf16(const _Float16* W, int K, int n0, int k0) {
  int lane = threadIdx.x & 31;
  int n = n0 + (lane & 15);
  int k = k0 + (lane >> 4) * 8;
  const v8h* p = (const v8h*)(W + (size_t)n * K + k);
  union { v16h v; v8h x[2]; } u;
  u.x[0] = p[0];
  u.x[1] = p[2];
  return u.v;
}

__device__ __forceinline__ float sigmoidf(float x) {
  return 1.0f / (1.0f + expf(-x));
}

// ---------------- generic WMMA GEMM + optional tanh ----------------
// out[m][n] = act( sum_k A[m][k] * W[n][k] + bias[n] ), out stored f16.
// Block = 8 waves; wave owns MI 16-row groups; B-frags cached per N-tile and
// reused across MI; A-frags register-resident when K*MI small enough.
template<int NT, int KT, int MI, int ACTF, bool AF32>
__global__ __launch_bounds__(256) void enc_gemm(const void* __restrict__ A,
                                                const _Float16* __restrict__ W,
                                                const float* __restrict__ bias,
                                                _Float16* __restrict__ out) {
  constexpr int K = KT * 32;
  constexpr int N = NT * 16;
  const int wave = threadIdx.x >> 5;
  const int lane = threadIdx.x & 31;
  const int col = lane & 15, half = lane >> 4;
  const int mbase = (blockIdx.x * 8 + wave) * (16 * MI);

  constexpr bool CACHE_A = (KT * MI <= 16);   // up to 128 VGPRs of A
  v16h afr[CACHE_A ? MI * KT : 1];
  if constexpr (CACHE_A) {
#pragma unroll
    for (int mi = 0; mi < MI; ++mi)
#pragma unroll
      for (int kt = 0; kt < KT; ++kt)
        afr[mi * KT + kt] = load_a_g<AF32>(A, K, mbase + mi * 16, kt * 32);
  }

  for (int nt = 0; nt < NT; ++nt) {
    const int n0 = nt * 16;
    v16h bfr[KT];
#pragma unroll
    for (int kt = 0; kt < KT; ++kt) bfr[kt] = load_b_gf16(W, K, n0, kt * 32);
    const float bv = bias[n0 + col];
#pragma unroll
    for (int mi = 0; mi < MI; ++mi) {
      v8f acc;
#pragma unroll
      for (int r = 0; r < 8; ++r) acc[r] = bv;
#pragma unroll
      for (int kt = 0; kt < KT; ++kt) {
        v16h a;
        if constexpr (CACHE_A) a = afr[mi * KT + kt];
        else                   a = load_a_g<AF32>(A, K, mbase + mi * 16, kt * 32);
        acc = wmma_f16(a, bfr[kt], acc);
      }
#pragma unroll
      for (int r = 0; r < 8; ++r) {
        float vo = acc[r];
        if (ACTF) vo = tanhf(vo);
        out[(size_t)(mbase + mi * 16 + r + 8 * half) * N + n0 + col] = (_Float16)vo;
      }
    }
  }
}

// ---------------- fp32 -> fp16 convert ----------------
__global__ void cvt_f16(const float* __restrict__ src, _Float16* __restrict__ dst, int n) {
  int i = blockIdx.x * 256 + threadIdx.x;
  if (i < n) dst[i] = (_Float16)src[i];
}

// ---------------- persistent LSTM + actor head ----------------
// 64 blocks x 256 threads; block owns 16 batch rows; loops t = 0..255.
// Per step: g = xg[t] (has bih) + bhh + h @ Whh^T via WMMA; gates; head fused.
// Whh B-fragments are loop-invariant -> compiler keeps them register-resident.
__global__ __launch_bounds__(256) void lstm_head(
    const _Float16* __restrict__ xg, const _Float16* __restrict__ whh16,
    const float* __restrict__ done, const float* __restrict__ h0,
    const float* __restrict__ c0, const float* __restrict__ bhh,
    const float* __restrict__ Wmean, const float* __restrict__ bmean,
    const float* __restrict__ logstd, const float* __restrict__ action,
    const float* __restrict__ z, float* __restrict__ out) {
  __shared__ float    sG[16 * 512];   // gate accumulators (cross-wave exchange)
  __shared__ _Float16 sH[16 * 128];   // hidden (A matrix for WMMA)
  __shared__ float    sC[16 * 128];   // cell state
  __shared__ float    sWm[8 * 128];   // Wmean
  __shared__ float    sBhh[512];
  __shared__ float    sD[16];
  __shared__ float    sLP[16 * 8];
  __shared__ float    sBm[8], sLs[8], sStd[8];
  __shared__ float    sEnt;

  const int tid = threadIdx.x;
  const int b0 = blockIdx.x * 16;
  const int wave = tid >> 5, lane = tid & 31;
  const int col = lane & 15, half = lane >> 4;

  for (int i = tid; i < 16 * 128; i += 256) {
    int m = i >> 7, j = i & 127;
    sH[i] = (_Float16)h0[(size_t)(b0 + m) * HL + j];
    sC[i] = c0[(size_t)(b0 + m) * HL + j];
  }
  for (int i = tid; i < 8 * 128; i += 256) sWm[i] = Wmean[i];
  for (int i = tid; i < 512; i += 256) sBhh[i] = bhh[i];
  if (tid < 8) {
    sBm[tid] = bmean[tid];
    float l = logstd[tid];
    sLs[tid] = l;
    sStd[tid] = expf(l);
  }
  __syncthreads();
  if (tid == 0) {
    float e = 0.f;
    for (int a = 0; a < 8; ++a) e += 0.5f + 0.5f * LOG2PI + sLs[a];
    sEnt = e;
  }
  __syncthreads();

  for (int t = 0; t < TT; ++t) {
    const size_t rowbase = (size_t)t * BB + b0;
    // prefetch next step's 16KB xg tile while this step computes
    if (t + 1 < TT) {
      const char* nxt = (const char*)(xg + ((size_t)(t + 1) * BB + b0) * 512);
      __builtin_prefetch(nxt + (size_t)tid * 64, 0, 3);
    }
    if (tid < 16) sD[tid] = 1.0f - done[rowbase + tid];
    __syncthreads();
    // reset h,c where done (before GEMM, matching reference)
    for (int i = tid; i < 2048; i += 256) {
      float mfac = sD[i >> 7];
      sH[i] = (_Float16)((float)sH[i] * mfac);
      sC[i] *= mfac;
    }
    __syncthreads();

    // g = xg + bhh + h @ Whh^T : each wave owns 4 N-tiles (4*16 of 512)
#pragma unroll
    for (int q = 0; q < 4; ++q) {
      const int n0 = (wave * 4 + q) * 16;
      v8f acc;
      const _Float16* xrow = xg + rowbase * 512 + n0 + col;
      const float bb = sBhh[n0 + col];
#pragma unroll
      for (int r = 0; r < 8; ++r)
        acc[r] = (float)xrow[(size_t)(r + 8 * half) * 512] + bb;
#pragma unroll
      for (int kt = 0; kt < 4; ++kt) {
        v16h a = load_a_g<false>(sH, HL, 0, kt * 32);
        v16h b = load_b_gf16(whh16, HL, n0, kt * 32);
        acc = wmma_f16(a, b, acc);
      }
#pragma unroll
      for (int r = 0; r < 8; ++r) sG[(r + 8 * half) * 512 + n0 + col] = acc[r];
    }
    __syncthreads();

    // gate nonlinearity + state update (torch order i, f, g, o)
    for (int i = tid; i < 2048; i += 256) {
      int m = i >> 7, j = i & 127;
      float gi = sG[m * 512 + j];
      float gf = sG[m * 512 + 128 + j];
      float gg = sG[m * 512 + 256 + j];
      float go = sG[m * 512 + 384 + j];
      float c = sigmoidf(gf) * sC[i] + sigmoidf(gi) * tanhf(gg);
      float h = sigmoidf(go) * tanhf(c);
      sC[i] = c;
      sH[i] = (_Float16)h;
    }
    __syncthreads();

    // actor head: mean = h @ Wmean^T + bmean + z ; logprob/entropy per row
    if (tid < 128) {
      int m = tid >> 3, a = tid & 7;
      float dot = 0.f;
#pragma unroll 16
      for (int j = 0; j < 128; ++j) dot += (float)sH[m * 128 + j] * sWm[a * 128 + j];
      size_t row = rowbase + m;
      float mean = dot + sBm[a] + z[row * ACTD + a];
      float diff = (action[row * ACTD + a] - mean) / sStd[a];
      sLP[tid] = -0.5f * diff * diff - sLs[a] - 0.5f * LOG2PI;
    }
    __syncthreads();
    if (tid < 16) {
      float s = 0.f;
#pragma unroll
      for (int a = 0; a < 8; ++a) s += sLP[tid * 8 + a];
      out[OUT_LP + rowbase + tid]  = s;
      out[OUT_ENT + rowbase + tid] = sEnt;
    }
    __syncthreads();  // protect sH/sC before next step's masking
  }

  // final hT, cT
  for (int i = tid; i < 2048; i += 256) {
    int m = i >> 7, j = i & 127;
    out[OUT_H + (size_t)(b0 + m) * HL + j] = (float)sH[m * 128 + j];
    out[OUT_C + (size_t)(b0 + m) * HL + j] = sC[i];
  }
}

// ---------------- launch ----------------
extern "C" void kernel_launch(void* const* d_in, const int* in_sizes, int n_in,
                              void* d_out, int out_size, void* d_ws, size_t ws_size,
                              hipStream_t stream) {
  const float* state  = (const float*)d_in[0];
  const float* done   = (const float*)d_in[1];
  const float* h0     = (const float*)d_in[2];
  const float* c0     = (const float*)d_in[3];
  const float* action = (const float*)d_in[4];
  const float* z      = (const float*)d_in[5];
  const float* W1     = (const float*)d_in[6];
  const float* b1     = (const float*)d_in[7];
  const float* W2     = (const float*)d_in[8];
  const float* b2     = (const float*)d_in[9];
  const float* Wih    = (const float*)d_in[10];
  const float* Whh    = (const float*)d_in[11];
  const float* bih    = (const float*)d_in[12];
  const float* bhh    = (const float*)d_in[13];
  const float* Wmean  = (const float*)d_in[14];
  const float* bmean  = (const float*)d_in[15];
  const float* logstd = (const float*)d_in[16];
  float* out = (float*)d_out;

  // workspace layout (f16): region0 = a1 (then reused as xg) | x2 | f16 weights
  char* ws = (char*)d_ws;
  size_t o = 0;
  _Float16* reg0  = (_Float16*)(ws + o); o += (size_t)TBN * H1 * 2;       // 268MB
  _Float16* x2    = (_Float16*)(ws + o); o += (size_t)TBN * H2 * 2;       // 134MB
  _Float16* whh16 = (_Float16*)(ws + o); o += (size_t)H1 * HL * 2;
  _Float16* w1_16 = (_Float16*)(ws + o); o += (size_t)H1 * OBS * 2;
  _Float16* w2_16 = (_Float16*)(ws + o); o += (size_t)H2 * H1 * 2;
  _Float16* wih16 = (_Float16*)(ws + o); o += (size_t)H1 * H2 * 2;
  _Float16* a1 = reg0;  // a1 is dead before xg is produced
  _Float16* xg = reg0;

  // action passthrough
  hipMemcpyAsync(out, action, (size_t)TBN * ACTD * sizeof(float),
                 hipMemcpyDeviceToDevice, stream);

  // one-time f32 -> f16 weight conversion (all L2-resident afterwards)
  cvt_f16<<<(H1 * HL  + 255) / 256, 256, 0, stream>>>(Whh, whh16, H1 * HL);
  cvt_f16<<<(H1 * OBS + 255) / 256, 256, 0, stream>>>(W1,  w1_16, H1 * OBS);
  cvt_f16<<<(H2 * H1  + 255) / 256, 256, 0, stream>>>(W2,  w2_16, H2 * H1);
  cvt_f16<<<(H1 * H2  + 255) / 256, 256, 0, stream>>>(Wih, wih16, H1 * H2);

  // encoder stage 1: a1 = tanh(state @ W1^T + b1)   [TB,128]x[128,512]
  enc_gemm<H1 / 16, OBS / 32, 2, 1, true><<<TBN / 256, 256, 0, stream>>>(state, w1_16, b1, a1);
  // encoder stage 2: x2 = tanh(a1 @ W2^T + b2)      [TB,512]x[512,256]
  enc_gemm<H2 / 16, H1 / 32, 2, 1, false><<<TBN / 256, 256, 0, stream>>>(a1, w2_16, b2, x2);
  // hoisted input GEMM: xg = x2 @ Wih^T + bih       [TB,256]x[256,512]
  enc_gemm<(4 * HL) / 16, H2 / 32, 2, 0, false><<<TBN / 256, 256, 0, stream>>>(x2, wih16, bih, xg);

  // sequential LSTM + fused actor head / logprob / entropy / hT / cT
  lstm_head<<<BB / 16, 256, 0, stream>>>(xg, whh16, done, h0, c0, bhh, Wmean,
                                         bmean, logstd, action, z, out);
}
